// AutoEncoder_11768210391563
// MI455X (gfx1250) — matrix operand order repaired and checked by reading, not compiled, not run on hardware
//
#include <hip/hip_runtime.h>
#include <hip/hip_bf16.h>

// ---------------- problem constants ----------------
#define Vn    8192
#define Bc    1024
#define INDIM 20000
#define OUTD  64
#define NNZ   (Vn * 32)
#define FC1FIN 3072   // F * V/P = 3 * 1024

typedef __attribute__((ext_vector_type(16))) __bf16 v16bf;
typedef __attribute__((ext_vector_type(8)))  __bf16 v8bf;
typedef __attribute__((ext_vector_type(8)))  float  v8f;

// ---------------- helpers ----------------
__device__ __forceinline__ __bf16 f2bf(float f) {
  unsigned u = __builtin_bit_cast(unsigned, f);
  u += 0x7fffu + ((u >> 16) & 1u);           // round-to-nearest-even
  unsigned short s = (unsigned short)(u >> 16);
  return __builtin_bit_cast(__bf16, s);
}

__device__ __forceinline__ float apply_act(float v, int ACT) {
  if (ACT == 1) return fmaxf(v, 0.0f);                       // relu
  if (ACT == 2) return v > 0.0f ? v : expm1f(v);             // celu(alpha=1)
  if (ACT == 3) return fminf(fmaxf(v, 0.0f), 1.0f);          // hardtanh(0,1)
  return v;
}

// ---------------- elementwise kernels ----------------
__global__ void zero_kernel(float* p, size_t n) {
  size_t i = (size_t)blockIdx.x * blockDim.x + threadIdx.x;
  size_t stride = (size_t)gridDim.x * blockDim.x;
  for (; i < n; i += stride) p[i] = 0.0f;
}

__global__ void cvt_bf16_kernel(const float* __restrict__ in, __bf16* __restrict__ out, size_t n) {
  size_t i = (size_t)blockIdx.x * blockDim.x + threadIdx.x;
  if (i < n) out[i] = f2bf(in[i]);
}

// x2 = 2*x2 - x0
__global__ void fixup_kernel(float* __restrict__ x2, const float* __restrict__ x0, size_t n) {
  size_t i = (size_t)blockIdx.x * blockDim.x + threadIdx.x;
  if (i < n) x2[i] = 2.0f * x2[i] - x0[i];
}

// c = bf16(0.5*(a+b))
__global__ void avg_bf_kernel(const float* __restrict__ a, const float* __restrict__ b,
                              __bf16* __restrict__ c, size_t n) {
  size_t i = (size_t)blockIdx.x * blockDim.x + threadIdx.x;
  if (i < n) c[i] = f2bf(0.5f * (a[i] + b[i]));
}

// ---------------- transpose: in[R,C] -> out[C,R]  (fp32) ----------------
__global__ void transpose_kernel(const float* __restrict__ in, float* __restrict__ out,
                                 int R, int C) {
  __shared__ float tile[32][33];
  int c0 = blockIdx.x * 32, r0 = blockIdx.y * 32;
  int tx = threadIdx.x, ty = threadIdx.y;   // 32 x 8
  #pragma unroll
  for (int i = ty; i < 32; i += 8)
    tile[i][tx] = in[(size_t)(r0 + i) * C + c0 + tx];
  __syncthreads();
  #pragma unroll
  for (int i = ty; i < 32; i += 8)
    out[(size_t)(c0 + i) * R + r0 + tx] = tile[tx][i];
}

// ---------------- SpMM: y[r,:] += val * x[c,:]  (x,y are [V,B]) ----------------
__global__ void spmm_kernel(const int* __restrict__ rows, const int* __restrict__ cols,
                            const float* __restrict__ vals,
                            const float* __restrict__ x, float* __restrict__ y) {
  int e = blockIdx.x;
  int r = rows[e], c = cols[e];
  float v = vals[e];
  const float* xc = x + (size_t)c * Bc;
  float* yr = y + (size_t)r * Bc;
  for (int b = threadIdx.x; b < Bc; b += blockDim.x)
    atomicAdd(&yr[b], v * xc[b]);
}

// ---------------- Chebyshev combine + relu + max-pool(P=8), bf16 out ----------------
__global__ void cheb_pool_kernel(const float* __restrict__ x0, const float* __restrict__ x1,
                                 const float* __restrict__ x2, const float* __restrict__ w,
                                 const float* __restrict__ bias, __bf16* __restrict__ xg) {
  int b  = blockIdx.x * blockDim.x + threadIdx.x;   // batch, coalesced
  int vp = blockIdx.y;                              // pooled node
  float w00=w[0],w01=w[1],w02=w[2], w10=w[3],w11=w[4],w12=w[5], w20=w[6],w21=w[7],w22=w[8];
  float b0=bias[0], b1=bias[1], b2=bias[2];
  float m0=0.f, m1=0.f, m2=0.f;                     // relu>=0 so 0 init is exact
  #pragma unroll
  for (int p = 0; p < 8; ++p) {
    size_t i = (size_t)(vp * 8 + p) * Bc + b;
    float f0 = x0[i], f1 = x1[i], f2 = x2[i];
    m0 = fmaxf(m0, fmaxf(w00*f0 + w01*f1 + w02*f2 + b0, 0.f));
    m1 = fmaxf(m1, fmaxf(w10*f0 + w11*f1 + w12*f2 + b1, 0.f));
    m2 = fmaxf(m2, fmaxf(w20*f0 + w21*f1 + w22*f2 + b2, 0.f));
  }
  size_t o = (size_t)b * FC1FIN + (size_t)vp * 3;
  xg[o] = f2bf(m0); xg[o+1] = f2bf(m1); xg[o+2] = f2bf(m2);
}

// ---------------- unified WMMA GEMM (bf16 operands, fp32 accumulate) ----------------
// NTMODE=1: C[M,N] = act(A[M,K] @ Bm[N,K]^T + bias)   (Bm row-major [N,K], bf16)
// NTMODE=0: C[M,N] = act(A[M,K] @ Bm[K,N])            (Bm row-major [K,N], bf16)
// OUTBF=1: store bf16, else fp32. Block = 4 waves; block tile 64M x 32N;
// wave tile 16M x 32N (2 accumulators, A fragment reused). B tile staged in LDS.
template <int ACT, int NTMODE, int OUTBF>
__global__ void gemm_wmma(const __bf16* __restrict__ A, const __bf16* __restrict__ Bm,
                          const float* __restrict__ bias, void* __restrict__ Cout,
                          int M, int N, int K) {
  __shared__ __align__(16) __bf16 ldsB[32][40];   // [n][k], 80B row stride, conflict-free
  const int lane = threadIdx.x;            // 0..31
  const int wave = threadIdx.y;            // 0..3
  const int tid  = wave * 32 + lane;       // 0..127
  const int tileM = (blockIdx.y * 4 + wave) * 16;
  const int tileN = blockIdx.x * 32;
  const bool active = tileM < M;           // wave-uniform; inactive waves still barrier
  const int h = lane >> 4;                 // K-half select
  const int r = lane & 15;
  int arowi = tileM + r; if (arowi >= M) arowi = M - 1;   // clamp for inactive waves
  const __bf16* arow = A + (size_t)arowi * K;
  v8f acc0 = {}, acc1 = {};

  for (int k0 = 0; k0 < K; k0 += 32) {
    __syncthreads();                       // protect previous iteration's LDS reads
    // ---- cooperative B tile stage: 32 N x 32 K bf16 -> LDS (pure copy) ----
    if (NTMODE) {
      int n = tid >> 2, ks = (tid & 3) * 8;
      const __bf16* src = Bm + (size_t)(tileN + n) * K + k0 + ks;
      if (k0 + 64 <= K) __builtin_prefetch(src + 64, 0, 1);   // global_prefetch_b8
      *(v8bf*)&ldsB[n][ks] = *(const v8bf*)src;
    } else {
      int n = tid & 31, kb = tid >> 5;     // coalesced over N, strided over K rows
      #pragma unroll
      for (int j = 0; j < 8; ++j) {
        int k = kb + j * 4;
        ldsB[n][k] = Bm[(size_t)(k0 + k) * N + tileN + n];
      }
    }
    __syncthreads();
    // ---- A fragment (ISA 16x32 layout): lanes 0-15 K{0-7,16-23}, 16-31 K{8-15,24-31}
    if (k0 + 64 <= K) __builtin_prefetch(arow + k0 + 64, 0, 1);
    v8bf alo = *(const v8bf*)(arow + k0 + 8 * h);
    v8bf ahi = *(const v8bf*)(arow + k0 + 16 + 8 * h);
    v16bf a = __builtin_shufflevector(alo, ahi, 0,1,2,3,4,5,6,7,8,9,10,11,12,13,14,15);
    // ---- B fragments: lane = N, lanes 0-15 hold K{0-7,16-23}, lanes 16-31 hold K{8-15,24-31}
    v8bf p0 = *(const v8bf*)&ldsB[r][8 * h];
    v8bf p1 = *(const v8bf*)&ldsB[r][16 + 8 * h];
    v8bf q0 = *(const v8bf*)&ldsB[16 + r][8 * h];
    v8bf q1 = *(const v8bf*)&ldsB[16 + r][16 + 8 * h];
    v16bf b0 = __builtin_shufflevector(p0, p1, 0,1,2,3,4,5,6,7,8,9,10,11,12,13,14,15);
    v16bf b1 = __builtin_shufflevector(q0, q1, 0,1,2,3,4,5,6,7,8,9,10,11,12,13,14,15);
    acc0 = __builtin_amdgcn_wmma_f32_16x16x32_bf16(false, a, false, b0,
                                                   (short)0, acc0, false, false);
    acc1 = __builtin_amdgcn_wmma_f32_16x16x32_bf16(false, a, false, b1,
                                                   (short)0, acc1, false, false);
  }

  if (active) {
    const int col0 = tileN + r, col1 = tileN + 16 + r;
    const float bv0 = bias ? bias[col0] : 0.0f;
    const float bv1 = bias ? bias[col1] : 0.0f;
    float*  Cf = (float*)Cout;
    __bf16* Cb = (__bf16*)Cout;
    #pragma unroll
    for (int i = 0; i < 8; ++i) {
      int row = tileM + i + 8 * h;         // C/D layout: VGPR i -> M = i + 8*(lane>>4)
      float v0 = apply_act(acc0[i] + bv0, ACT);
      float v1 = apply_act(acc1[i] + bv1, ACT);
      if (OUTBF) { Cb[(size_t)row * N + col0] = f2bf(v0); Cb[(size_t)row * N + col1] = f2bf(v1); }
      else       { Cf[(size_t)row * N + col0] = v0;       Cf[(size_t)row * N + col1] = v1; }
    }
  }
}

// ---------------- softmax over 64 columns; fp32 + bf16 dual store ----------------
__global__ void softmax64_kernel(const float* __restrict__ in, float* __restrict__ out,
                                 __bf16* __restrict__ outb) {
  __shared__ float sh[64];
  int row = blockIdx.x, t = threadIdx.x;
  float v = in[(size_t)row * 64 + t];
  sh[t] = v; __syncthreads();
  for (int s = 32; s > 0; s >>= 1) { if (t < s) sh[t] = fmaxf(sh[t], sh[t + s]); __syncthreads(); }
  float mx = sh[0]; __syncthreads();
  float e = expf(v - mx);
  sh[t] = e; __syncthreads();
  for (int s = 32; s > 0; s >>= 1) { if (t < s) sh[t] += sh[t + s]; __syncthreads(); }
  float p = e / sh[0];
  out[(size_t)row * 64 + t] = p;
  outb[(size_t)row * 64 + t] = f2bf(p);
}

// ---------------- host helpers ----------------
static inline dim3 ggrid(int M, int N) { return dim3(N / 32, (M + 63) / 64); }
static inline void cvt(const float* in, __bf16* out, size_t n, hipStream_t s) {
  cvt_bf16_kernel<<<(unsigned)((n + 255) / 256), 256, 0, s>>>(in, out, n);
}

extern "C" void kernel_launch(void* const* d_in, const int* in_sizes, int n_in,
                              void* d_out, int out_size, void* d_ws, size_t ws_size,
                              hipStream_t stream) {
  const float* x_gcn = (const float*)d_in[0];
  const float* x_nn  = (const float*)d_in[1];
  const int*   L_rows = (const int*)d_in[2];
  const int*   L_cols = (const int*)d_in[3];
  const float* L_vals = (const float*)d_in[4];
  const float* cl1_w = (const float*)d_in[5];  const float* cl1_b = (const float*)d_in[6];
  const float* fc_w[3] = {(const float*)d_in[7], (const float*)d_in[9], (const float*)d_in[11]};
  const float* fc_b[3] = {(const float*)d_in[8], (const float*)d_in[10], (const float*)d_in[12]};
  const float* enc_w[5] = {(const float*)d_in[13], (const float*)d_in[15], (const float*)d_in[17],
                           (const float*)d_in[19], (const float*)d_in[21]};
  const float* enc_b[5] = {(const float*)d_in[14], (const float*)d_in[16], (const float*)d_in[18],
                           (const float*)d_in[20], (const float*)d_in[22]};
  const float* dec_w[5] = {(const float*)d_in[23], (const float*)d_in[24], (const float*)d_in[25],
                           (const float*)d_in[26], (const float*)d_in[27]};
  const float* sum_w = (const float*)d_in[28]; const float* sum_b = (const float*)d_in[29];

  float* out     = (float*)d_out;
  float* x_recon = out;                                   // [1024, 20000]
  float* z       = out + (size_t)Bc * INDIM;              // [1024, 64]
  float* sig_out = z + (size_t)Bc * OUTD;                 // [64, 20000]

  // ---- workspace ----
  float* ws = (float*)d_ws;
  const size_t VB = (size_t)Vn * Bc;                      // 8,388,608
  // phase 1 (fp32 graph buffers; dead after cheb):
  float* x0T = ws;
  float* x1b = ws + VB;
  float* x2b = ws + 2 * VB;
  // persistent bf16 xg + small fp32 region, after the graph buffers:
  __bf16* xg_bf = (__bf16*)(ws + 3 * VB);                 // 3,145,728 bf16
  float* f32r   = ws + 3 * VB + 1572864;                  // = after xg_bf
  float* gfeat  = f32r;                                   // 1024*32
  float* nfeat  = gfeat + 32768;                          // 1024*32
  float* logit  = nfeat + 32768;                          // 1024*64
  float* sigT   = logit + 65536;                          // [20000,64] fp32 (clipped)
  // phase 2 bf16 mirror region: aliases x0T/x1/x2 (only used after cheb)
  __bf16* bw = (__bf16*)d_ws;
  size_t o = 0;
  auto take = [&](size_t n) { __bf16* p = bw + o; o += n; return p; };
  __bf16* x_nn_bf  = take((size_t)Bc * INDIM);        // 20,480,000
  __bf16* e0w_bf   = take((size_t)512 * INDIM);       // 10,240,000
  __bf16* d4w_bf   = take((size_t)INDIM * 512);       // 10,240,000
  __bf16* f1w_bf   = take((size_t)512 * FC1FIN);      // 1,572,864
  __bf16* sigT_bf  = take((size_t)INDIM * OUTD);      // 1,280,000
  __bf16* g1_bf    = take(1024 * 512);
  __bf16* h0_bf    = take(1024 * 512);
  __bf16* g2_bf    = take(1024 * 256);
  __bf16* h1_bf    = take(1024 * 256);
  __bf16* f2w_bf   = take(256 * 512);
  __bf16* e1w_bf   = take(256 * 512);
  __bf16* d3w_bf   = take(512 * 256);
  __bf16* h2_bf    = take(1024 * 128);
  __bf16* h3_bf    = take(1024 * 64);
  __bf16* z_bf     = take(1024 * 64);
  __bf16* e2w_bf   = take(128 * 256);
  __bf16* d2w_bf   = take(256 * 128);
  __bf16* R3_bf    = take(512 * 64);
  __bf16* f3w_bf   = take(32 * 256);
  __bf16* R2_bf    = take(256 * 64);
  __bf16* R1_bf    = take(128 * 64);
  __bf16* d1w_bf   = take(128 * 64);
  __bf16* d0w_bf   = take(64 * 64);
  __bf16* e3w_bf   = take(64 * 128);
  __bf16* e4w_bf   = take(32 * 64);
  __bf16* sw_bf    = take(64 * 32);
  __bf16* xs_bf    = take(1024 * 32);                 // total ~46.2M bf16 < 50.3M capacity

  dim3 blk(32, 4);

  // ---- phase 1: graph conv ----
  zero_kernel<<<4096, 256, 0, stream>>>(x1b, 2 * VB);
  transpose_kernel<<<dim3(Vn / 32, Bc / 32), dim3(32, 8), 0, stream>>>(x_gcn, x0T, Bc, Vn);
  spmm_kernel<<<NNZ, 256, 0, stream>>>(L_rows, L_cols, L_vals, x0T, x1b);
  spmm_kernel<<<NNZ, 256, 0, stream>>>(L_rows, L_cols, L_vals, x1b, x2b);
  fixup_kernel<<<(unsigned)(VB / 256), 256, 0, stream>>>(x2b, x0T, VB);
  cheb_pool_kernel<<<dim3(Bc / 256, Vn / 8), 256, 0, stream>>>(x0T, x1b, x2b, cl1_w, cl1_b, xg_bf);

  // ---- phase 2: one-pass fp32 -> bf16 conversions (graph buffers now dead) ----
  cvt(x_nn, x_nn_bf, (size_t)Bc * INDIM, stream);
  cvt(enc_w[0], e0w_bf, (size_t)512 * INDIM, stream);
  cvt(enc_w[1], e1w_bf, 256 * 512, stream);
  cvt(enc_w[2], e2w_bf, 128 * 256, stream);
  cvt(enc_w[3], e3w_bf, 64 * 128, stream);
  cvt(enc_w[4], e4w_bf, 32 * 64, stream);
  cvt(dec_w[0], d0w_bf, 64 * 64, stream);
  cvt(dec_w[1], d1w_bf, 128 * 64, stream);
  cvt(dec_w[2], d2w_bf, 256 * 128, stream);
  cvt(dec_w[3], d3w_bf, 512 * 256, stream);
  cvt(dec_w[4], d4w_bf, (size_t)INDIM * 512, stream);
  cvt(fc_w[0], f1w_bf, (size_t)512 * FC1FIN, stream);
  cvt(fc_w[1], f2w_bf, 256 * 512, stream);
  cvt(fc_w[2], f3w_bf, 32 * 256, stream);
  cvt(sum_w, sw_bf, 64 * 32, stream);

  // ---- GCN head: fc1(relu) fc2(relu) fc3 ----
  gemm_wmma<1,1,1><<<ggrid(Bc,512), blk, 0, stream>>>(xg_bf, f1w_bf, fc_b[0], g1_bf, Bc, 512, FC1FIN);
  gemm_wmma<1,1,1><<<ggrid(Bc,256), blk, 0, stream>>>(g1_bf, f2w_bf, fc_b[1], g2_bf, Bc, 256, 512);
  gemm_wmma<0,1,0><<<ggrid(Bc,32),  blk, 0, stream>>>(g2_bf, f3w_bf, fc_b[2], gfeat, Bc, 32, 256);

  // ---- sig^T = D4 @ D3 @ D2 @ D1 @ D0 (right-to-left), clip01 on last ----
  gemm_wmma<0,0,1><<<ggrid(128,64), blk, 0, stream>>>(d1w_bf, d0w_bf, nullptr, R1_bf, 128, 64, 64);
  gemm_wmma<0,0,1><<<ggrid(256,64), blk, 0, stream>>>(d2w_bf, R1_bf, nullptr, R2_bf, 256, 64, 128);
  gemm_wmma<0,0,1><<<ggrid(512,64), blk, 0, stream>>>(d3w_bf, R2_bf, nullptr, R3_bf, 512, 64, 256);
  gemm_wmma<3,0,0><<<ggrid(INDIM,64), blk, 0, stream>>>(d4w_bf, R3_bf, nullptr, sigT, INDIM, 64, 512);
  cvt(sigT, sigT_bf, (size_t)INDIM * OUTD, stream);

  // ---- encoder: celu x4, linear last ----
  gemm_wmma<2,1,1><<<ggrid(Bc,512), blk, 0, stream>>>(x_nn_bf, e0w_bf, enc_b[0], h0_bf, Bc, 512, INDIM);
  gemm_wmma<2,1,1><<<ggrid(Bc,256), blk, 0, stream>>>(h0_bf, e1w_bf, enc_b[1], h1_bf, Bc, 256, 512);
  gemm_wmma<2,1,1><<<ggrid(Bc,128), blk, 0, stream>>>(h1_bf, e2w_bf, enc_b[2], h2_bf, Bc, 128, 256);
  gemm_wmma<2,1,1><<<ggrid(Bc,64),  blk, 0, stream>>>(h2_bf, e3w_bf, enc_b[3], h3_bf, Bc, 64, 128);
  gemm_wmma<0,1,0><<<ggrid(Bc,32),  blk, 0, stream>>>(h3_bf, e4w_bf, enc_b[4], nfeat, Bc, 32, 64);

  // ---- fuse + softmax -> z (fp32 out + bf16 for recon) ----
  avg_bf_kernel<<<(Bc * 32) / 256, 256, 0, stream>>>(nfeat, gfeat, xs_bf, (size_t)Bc * 32);
  gemm_wmma<0,1,0><<<ggrid(Bc,64), blk, 0, stream>>>(xs_bf, sw_bf, sum_b, logit, Bc, 64, 32);
  softmax64_kernel<<<Bc, 64, 0, stream>>>(logit, z, z_bf);

  // ---- x_recon = z @ sig = z @ (sigT)^T ----
  gemm_wmma<0,1,0><<<ggrid(Bc,INDIM), blk, 0, stream>>>(z_bf, sigT_bf, nullptr, x_recon, Bc, INDIM, 64);

  // ---- sig output [64, 20000] = transpose(sigT) ----
  transpose_kernel<<<dim3(OUTD / 32, INDIM / 32), dim3(32, 8), 0, stream>>>(sigT, sig_out, INDIM, OUTD);
}
